// SpatialAttn_86887188398862
// MI455X (gfx1250) — compile-verified, matrix operand order checked by reading, not run on hardware
//
#include <hip/hip_runtime.h>
#include <hip/hip_bf16.h>

typedef __attribute__((ext_vector_type(16))) __bf16 v16bf;
typedef __attribute__((ext_vector_type(8)))  __bf16 v8bf;
typedef __attribute__((ext_vector_type(4)))  __bf16 v4bf;
typedef __attribute__((ext_vector_type(8)))  float  v8f;
typedef __attribute__((ext_vector_type(4)))  int    i4;

#define DIM   512
#define NJ    22
#define NTOK  8192
#define RROWS (NTOK * NJ)      /* 180224 flat (n,j) rows */
#define LDA   520              /* padded LDS row stride in bf16 elems */
#define MATEL 262144           /* 512*512 elems per packed matrix */

#if defined(__has_builtin)
#if __has_builtin(__builtin_amdgcn_global_load_async_to_lds_b128)
#define USE_ASYNC_LDS 1
#endif
#endif

// -------------------------------------------------------------------------
// Kernel 0: convert weights fp32 -> bf16 in WMMA B-fragment order.
// B fragment (16-bit, 32x16): lane l: col N = l%16 ; element e (0..15):
//   K = 32*kt + (l<16 ? 0 : 8) + (e<8 ? e : e+8)
// Flat pack index: (((kt*32 + nt)*32 + lane)*16 + e)
// mats 0..2 = k_w, v_w, o_w applied as x@W^T  -> B[K][N] = W[N*512+K]
// mats 3+j  = q_projs[j] applied as x@W       -> B[K][N] = Q[j][K*512+N]
// -------------------------------------------------------------------------
__global__ __launch_bounds__(256) void pack_weights(
    const float* __restrict__ kw, const float* __restrict__ vw,
    const float* __restrict__ ow, const float* __restrict__ qp,
    __bf16* __restrict__ pack)
{
  size_t t = (size_t)blockIdx.x * 256 + threadIdx.x;   // 25*262144 total
  int e    = (int)(t & 15);
  int lane = (int)((t >> 4) & 31);
  int nt   = (int)((t >> 9) & 31);
  int kt   = (int)((t >> 14) & 15);
  int mat  = (int)(t >> 18);
  int K = kt * 32 + ((lane & 16) ? 8 : 0) + ((e < 8) ? e : e + 8);
  int N = nt * 16 + (lane & 15);
  float val;
  if (mat < 3) {
    const float* W = (mat == 0) ? kw : (mat == 1) ? vw : ow;
    val = W[(size_t)N * DIM + K];                       // transpose
  } else {
    val = qp[(size_t)(mat - 3) * MATEL + (size_t)K * DIM + N];
  }
  pack[t] = (__bf16)val;
}

// -------------------------------------------------------------------------
// GEMM helpers: stage fp32 tiles into bf16 LDS (padded rows), and load an
// A-fragment per the 16-bit A layout (two 16B runs per lane).
// -------------------------------------------------------------------------
__device__ __forceinline__ void stage_quad(__bf16* sA, int q, float4 v) {
  int row = q >> 7;               // 128 quads per 512-wide row
  int c4  = (q & 127) << 2;
  v4bf b;
  b.x = (__bf16)v.x; b.y = (__bf16)v.y; b.z = (__bf16)v.z; b.w = (__bf16)v.w;
  *(v4bf*)&sA[row * LDA + c4] = b;
}

__device__ __forceinline__ v16bf load_afrag(const __bf16* sA, int mrow, int lane, int kt) {
  const __bf16* ap = &sA[(mrow + (lane & 15)) * LDA + kt * 32 + ((lane & 16) ? 8 : 0)];
  v8bf lo = *(const v8bf*)ap;
  v8bf hi = *(const v8bf*)(ap + 16);
  return __builtin_shufflevector(lo, hi, 0,1,2,3,4,5,6,7,8,9,10,11,12,13,14,15);
}

// -------------------------------------------------------------------------
// Kernel 1: K,V projection.  64 flat rows per block (contiguous in x),
// 512 threads = 16 waves.  Waves 0-7 -> K, 8-15 -> V; each wave owns a
// 64-col slice (4 n-subtiles) and 4 M-tiles: B fragments reused 4x.
// -------------------------------------------------------------------------
__global__ __launch_bounds__(512) void kv_proj(
    const float* __restrict__ x, const __bf16* __restrict__ pack,
    const float* __restrict__ kb, const float* __restrict__ vb,
    __bf16* __restrict__ wK, __bf16* __restrict__ wV)
{
  __shared__ __bf16 sA[64 * LDA];
  const int tid = threadIdx.x;
  const size_t r0 = (size_t)blockIdx.x * 64;
  const float* src = x + r0 * DIM;
  #pragma unroll
  for (int i = 0; i < 16; ++i) {
    int q = tid + i * 512;                      // 8192 quads = 64x512
    stage_quad(sA, q, ((const float4*)src)[q]);
  }
  __syncthreads();

  const int lane = tid & 31, wave = tid >> 5;
  const bool isV = wave >= 8;
  const __bf16* pk = pack + (isV ? MATEL : 0);
  const int nBase = (wave & 7) * 4;
  v8f acc[16] = {};
  for (int kt = 0; kt < 16; ++kt) {
    v16bf b[4];
    #pragma unroll
    for (int s = 0; s < 4; ++s)
      b[s] = *(const v16bf*)&pk[(size_t)((kt * 32 + nBase + s) * 32 + lane) * 16];
    #pragma unroll
    for (int m = 0; m < 4; ++m) {
      v16bf a = load_afrag(sA, m * 16, lane, kt);
      #pragma unroll
      for (int s = 0; s < 4; ++s)
        acc[m * 4 + s] = __builtin_amdgcn_wmma_f32_16x16x32_bf16(
            false, a, false, b[s], (short)0, acc[m * 4 + s], false, false);
    }
  }
  const float* bias = isV ? vb : kb;
  __bf16* outp = isV ? wV : wK;
  const int rB = (lane & 16) ? 8 : 0;
  for (int s = 0; s < 4; ++s) {
    int col = (nBase + s) * 16 + (lane & 15);
    float bv = bias[col];
    for (int m = 0; m < 4; ++m) {
      #pragma unroll
      for (int i = 0; i < 8; ++i) {
        float v = acc[m * 4 + s][i] + bv;
        outp[(r0 + m * 16 + rB + i) * DIM + col] = (__bf16)v;
      }
    }
  }
}

// -------------------------------------------------------------------------
// Kernel 2: per-joint Q projection.  Block = (64 tokens) x (joint j),
// 16 waves x 2 n-subtiles x 4 M-tiles.  1/sqrt(hd)=0.125 folded in.
// -------------------------------------------------------------------------
__global__ __launch_bounds__(512) void q_proj(
    const float* __restrict__ x, const __bf16* __restrict__ qpack,
    __bf16* __restrict__ wQ)
{
  __shared__ __bf16 sA[64 * LDA];
  const int tid = threadIdx.x;
  const int j = blockIdx.y;
  const size_t n0 = (size_t)blockIdx.x * 64;
  #pragma unroll
  for (int i = 0; i < 16; ++i) {
    int q = tid + i * 512;
    int row = q >> 7;
    int c4  = (q & 127) << 2;
    float4 v = *(const float4*)(x + ((n0 + row) * NJ + j) * DIM + c4);
    stage_quad(sA, q, v);
  }
  __syncthreads();

  const int lane = tid & 31, wave = tid >> 5;
  const __bf16* pk = qpack + (size_t)j * MATEL;
  const int nBase = wave * 2;
  v8f acc[8] = {};
  for (int kt = 0; kt < 16; ++kt) {
    v16bf b[2];
    #pragma unroll
    for (int s = 0; s < 2; ++s)
      b[s] = *(const v16bf*)&pk[(size_t)((kt * 32 + nBase + s) * 32 + lane) * 16];
    #pragma unroll
    for (int m = 0; m < 4; ++m) {
      v16bf a = load_afrag(sA, m * 16, lane, kt);
      #pragma unroll
      for (int s = 0; s < 2; ++s)
        acc[m * 2 + s] = __builtin_amdgcn_wmma_f32_16x16x32_bf16(
            false, a, false, b[s], (short)0, acc[m * 2 + s], false, false);
    }
  }
  const int rB = (lane & 16) ? 8 : 0;
  for (int s = 0; s < 2; ++s) {
    int col = (nBase + s) * 16 + (lane & 15);
    for (int m = 0; m < 4; ++m) {
      #pragma unroll
      for (int i = 0; i < 8; ++i) {
        float v = acc[m * 2 + s][i] * 0.125f;        // fold 1/sqrt(64)
        size_t r = (n0 + m * 16 + rB + i) * NJ + j;
        wQ[r * DIM + col] = (__bf16)v;
      }
    }
  }
}

// -------------------------------------------------------------------------
// Kernel 3: masked attention per token; one wave per head.  K/V staged to
// LDS via async-to-LDS when available (ASYNCcnt path).  Context written
// fp32 straight into d_out (same [r,512] layout as the final output).
// -------------------------------------------------------------------------
__constant__ unsigned int ADJ[NJ] = {
  0x0000000Fu,0x00000013u,0x00000025u,0x00000049u,0x00000092u,0x00000124u,
  0x00000248u,0x00000490u,0x00000920u,0x00007240u,0x00000480u,0x00000900u,
  0x00009200u,0x00012200u,0x00024200u,0x00009000u,0x00052000u,0x000A4000u,
  0x00150000u,0x002A0000u,0x00140000u,0x00280000u };

__global__ __launch_bounds__(256) void attn(
    const __bf16* __restrict__ wQ, const __bf16* __restrict__ wK,
    const __bf16* __restrict__ wV, float* __restrict__ out)
{
  __shared__ __align__(16) __bf16 sK[NJ * DIM];
  __shared__ __align__(16) __bf16 sV[NJ * DIM];
  __shared__ float sS[8 * NJ * NJ];
  const int tid = threadIdx.x;
  const size_t base = (size_t)blockIdx.x * (NJ * DIM);
  const __bf16* gK = wK + base;
  const __bf16* gV = wV + base;
#ifdef USE_ASYNC_LDS
  for (int i = tid; i < (NJ * DIM) / 8; i += 256) {    // 1408 x b128
    __builtin_amdgcn_global_load_async_to_lds_b128(
        (__attribute__((address_space(1))) i4*)(void*)(gK + (size_t)i * 8),
        (__attribute__((address_space(3))) i4*)(void*)&sK[i * 8], 0, 0);
    __builtin_amdgcn_global_load_async_to_lds_b128(
        (__attribute__((address_space(1))) i4*)(void*)(gV + (size_t)i * 8),
        (__attribute__((address_space(3))) i4*)(void*)&sV[i * 8], 0, 0);
  }
#if __has_builtin(__builtin_amdgcn_s_wait_asynccnt)
  __builtin_amdgcn_s_wait_asynccnt(0);
#else
  asm volatile("s_wait_asynccnt 0x0" ::: "memory");
#endif
#else
  for (int i = tid; i < (NJ * DIM) / 8; i += 256) {    // 1408 x uint4
    ((uint4*)sK)[i] = ((const uint4*)gK)[i];
    ((uint4*)sV)[i] = ((const uint4*)gV)[i];
  }
#endif
  __syncthreads();

  const int lane = tid & 31, h = tid >> 5;
  for (int p = lane; p < NJ * NJ; p += 32) {
    int j1 = p / NJ, j2 = p - j1 * NJ;
    float sc = -__builtin_inff();
    if ((ADJ[j1] >> j2) & 1u) {
      const unsigned* qp = (const unsigned*)(wQ + base + (size_t)j1 * DIM + h * 64);
      const unsigned* kp = (const unsigned*)(sK + j2 * DIM + h * 64);
      float acc = 0.f;
      #pragma unroll 8
      for (int k = 0; k < 32; ++k) {
        unsigned qu = qp[k], ku = kp[k];
        acc += __uint_as_float((qu & 0xffffu) << 16) * __uint_as_float((ku & 0xffffu) << 16)
             + __uint_as_float(qu & 0xffff0000u)     * __uint_as_float(ku & 0xffff0000u);
      }
      sc = acc;                                  // scale already folded into Q
    }
    sS[h * (NJ * NJ) + p] = sc;
  }
  __syncthreads();

  if (tid < 8 * NJ) {                            // softmax: one (h, j1) per thread
    int hh = tid / NJ, j1 = tid - hh * NJ;
    float* row = &sS[hh * (NJ * NJ) + j1 * NJ];
    float m = row[0];
    for (int k = 1; k < NJ; ++k) m = fmaxf(m, row[k]);
    float ssum = 0.f;
    for (int k = 0; k < NJ; ++k) ssum += __expf(row[k] - m);
    float inv = 1.f / ssum;
    for (int k = 0; k < NJ; ++k) row[k] = __expf(row[k] - m) * inv;
  }
  __syncthreads();

  float* orow = out + base;                      // ctx in-place in d_out
  for (int j1 = 0; j1 < NJ; ++j1) {
    const float* arow = &sS[h * (NJ * NJ) + j1 * NJ];
    float a0 = 0.f, a1 = 0.f;
    for (int j2 = 0; j2 < NJ; ++j2) {
      float w = arow[j2];
      a0 += w * (float)sV[j2 * DIM + h * 64 + lane];
      a1 += w * (float)sV[j2 * DIM + h * 64 + 32 + lane];
    }
    orow[j1 * DIM + h * 64 + lane]      = a0;
    orow[j1 * DIM + h * 64 + 32 + lane] = a1;
  }
}

// -------------------------------------------------------------------------
// Kernel 4: output projection, in place on d_out.  Block owns 64 full rows:
// stage them (fp32 -> bf16 LDS), barrier, then overwrite with ctx@o_w^T + b.
// 16 waves x 2 n-subtiles x 4 M-tiles.
// -------------------------------------------------------------------------
__global__ __launch_bounds__(512) void out_proj(
    const __bf16* __restrict__ opack, const float* __restrict__ ob,
    float* __restrict__ out)
{
  __shared__ __bf16 sA[64 * LDA];
  const int tid = threadIdx.x;
  const size_t r0 = (size_t)blockIdx.x * 64;
  const float* src = out + r0 * DIM;
  #pragma unroll
  for (int i = 0; i < 16; ++i) {
    int q = tid + i * 512;
    stage_quad(sA, q, ((const float4*)src)[q]);
  }
  __syncthreads();

  const int lane = tid & 31, wave = tid >> 5;
  const int nBase = wave * 2;
  v8f acc[8] = {};
  for (int kt = 0; kt < 16; ++kt) {
    v16bf b[2];
    #pragma unroll
    for (int s = 0; s < 2; ++s)
      b[s] = *(const v16bf*)&opack[(size_t)((kt * 32 + nBase + s) * 32 + lane) * 16];
    #pragma unroll
    for (int m = 0; m < 4; ++m) {
      v16bf a = load_afrag(sA, m * 16, lane, kt);
      #pragma unroll
      for (int s = 0; s < 2; ++s)
        acc[m * 2 + s] = __builtin_amdgcn_wmma_f32_16x16x32_bf16(
            false, a, false, b[s], (short)0, acc[m * 2 + s], false, false);
    }
  }
  const int rB = (lane & 16) ? 8 : 0;
  for (int s = 0; s < 2; ++s) {
    int col = (nBase + s) * 16 + (lane & 15);
    float bv = ob[col];
    for (int m = 0; m < 4; ++m) {
      #pragma unroll
      for (int i = 0; i < 8; ++i)
        out[(r0 + m * 16 + rB + i) * DIM + col] = acc[m * 2 + s][i] + bv;
    }
  }
}

// -------------------------------------------------------------------------
extern "C" void kernel_launch(void* const* d_in, const int* in_sizes, int n_in,
                              void* d_out, int out_size, void* d_ws, size_t ws_size,
                              hipStream_t stream) {
  const float* x  = (const float*)d_in[0];
  const float* qp = (const float*)d_in[1];
  const float* kw = (const float*)d_in[2];
  const float* kb = (const float*)d_in[3];
  const float* vw = (const float*)d_in[4];
  const float* vb = (const float*)d_in[5];
  const float* ow = (const float*)d_in[6];
  const float* ob = (const float*)d_in[7];
  float* out = (float*)d_out;

  char* ws = (char*)d_ws;
  __bf16* pack = (__bf16*)ws;                      // 25 * 262144 bf16 = 12.5 MiB
  const size_t packB = (size_t)25 * MATEL * 2;
  const size_t big   = (size_t)RROWS * DIM * 2;    // 176 MiB each
  __bf16* wK = (__bf16*)(ws + packB);
  __bf16* wV = (__bf16*)(ws + packB + big);
  __bf16* wQ = (__bf16*)(ws + packB + 2 * big);

  pack_weights<<<25 * MATEL / 256, 256, 0, stream>>>(kw, vw, ow, qp, pack);
  kv_proj<<<RROWS / 64, 512, 0, stream>>>(x, pack, kb, vb, wK, wV);
  q_proj<<<dim3(NTOK / 64, NJ), 512, 0, stream>>>(x, pack + (size_t)3 * MATEL, wQ);
  attn<<<NTOK, 256, 0, stream>>>(wQ, wK, wV, out);
  out_proj<<<RROWS / 64, 512, 0, stream>>>(pack + (size_t)2 * MATEL, ob, out);
}
